// Multi_VAE_50594714747484
// MI455X (gfx1250) — matrix operand order repaired and checked by reading, not compiled, not run on hardware
//
#include <hip/hip_runtime.h>
#include <hip/hip_bf16.h>

// Multi-view VAE forward for MI455X (gfx1250, wave32, WMMA).
// One templated WMMA GEMM kernel: 256 threads (8 waves), 128x128 tile, BK=32,
// each wave = 2x4 v_wmma_f32_16x16x32_f16. Double-buffered LDS pipeline:
//   - A (f16 activations): TDM tensor_load_to_lds (TENSORcnt) when available,
//     else GLOBAL_LOAD_ASYNC_TO_LDS_B128 (ASYNCcnt)
//   - A (f32 x-inputs) and B (f32 weights): register-staged with f16 convert,
//     B transposed into LDS so each lane's fragment is two 16B runs of one row.
//
// d_in layout assumption (setup_inputs() insertion order, params flattened
// in declaration order): 0:x0 1:x1 2:w 3:eps, then per view v (base 4+10v):
//   +0 encW0[2048x1024] +1 encB0 +2 encW1[1024x512] +3 encB1
//   +4 decW0[256x512]   +5 decB0 +6 decW1[512x1024] +7 decB1
//   +8 decW2[1024x2048] +9 decB2
// then 24:Wm[512x256] 25:bm 26:Wv[512x256] 27:bv.
// d_out = [loss_rec, loss_kl, n_fea_v_tensor(8192x512x2 row-major)].

#define NROWS 8192
#define D_IN  2048
#define H1    1024
#define H2    512
#define LAT   256

// Async global->LDS path (proven to assemble in round 2).
#define VAE_USE_ASYNC 1
// TDM path: use the Tensor Data Mover for A tiles when the builtin exists.
#if defined(__has_builtin)
#  if __has_builtin(__builtin_amdgcn_tensor_load_to_lds)
#    define VAE_HAVE_TDM 1
#  endif
#endif
#ifndef VAE_HAVE_TDM
#  define VAE_HAVE_TDM 0
#endif

typedef _Float16     v16h  __attribute__((ext_vector_type(16)));
typedef _Float16     half8 __attribute__((ext_vector_type(8)));
typedef _Float16     half4 __attribute__((ext_vector_type(4)));
typedef float        v8f   __attribute__((ext_vector_type(8)));
typedef float        f32x4 __attribute__((ext_vector_type(4)));
typedef unsigned int u32x4 __attribute__((ext_vector_type(4)));
typedef int          i32x8 __attribute__((ext_vector_type(8)));
typedef int          i32x4 __attribute__((ext_vector_type(4)));

enum { EP_RELU_H16 = 0, EP_FEA = 1, EP_F32 = 2, EP_BCE = 3 };

struct GemmArgs {
    const void*  A;        // [M x K] row-major, f16 or f32 (AF32)
    const float* B;        // [K x N] row-major fp32 weights
    const float* bias;     // [N]
    int M, N, K;
    void*        out;      // f16 (RELU_H16) or f32 (FEA / F32)
    const float* wcol;     // w [M x 2] (FEA / BCE)
    int          v;        // view index (FEA / BCE)
    const float* target;   // x_v [M x N] fp32 (BCE)
    float*       partials; // per-block BCE partial sums (BCE)
};

__device__ __forceinline__ v16h pack16(half8 lo, half8 hi) {
    v16h r;
#pragma unroll
    for (int i = 0; i < 8; ++i) { r[i] = lo[i]; r[i + 8] = hi[i]; }
    return r;
}

// LDS row pitch of 40 halves (80B) keeps all fragment reads 16B-aligned.
// For TDM this is pad_interval = 16 DWORDs (64B row), pad_amount = 4 DWORDs.
#define APITCH 40
#define ATILE  (128 * APITCH)

template <int EPI, bool AF32>
__global__ __launch_bounds__(256) void vae_gemm_wmma(GemmArgs g) {
    const int tid  = threadIdx.x;
    const int lane = tid & 31;
    const int wave = tid >> 5;
    const int wm   = wave & 3;   // 4 wave rows  (32 rows each)
    const int wn   = wave >> 2;  // 2 wave cols  (64 cols each)
    const int bM   = blockIdx.y * 128;
    const int bN   = blockIdx.x * 128;
    const int gr   = lane & 15;  // lane-in-group
    const int grp  = lane >> 4;  // lane group 0/1

    __shared__ _Float16 Alds[2][ATILE];
    __shared__ _Float16 Blds[2][ATILE];

    v8f acc[2][4];
#pragma unroll
    for (int i = 0; i < 2; ++i)
#pragma unroll
        for (int j = 0; j < 4; ++j)
#pragma unroll
            for (int r = 0; r < 8; ++r) acc[i][j][r] = 0.0f;

    // ---- staging helpers -------------------------------------------------
    // A tile (f16 path): 128 rows x 32 halves, row pitch 40 halves in LDS.
    auto stageA16 = [&](int kb, int buf) {
        const _Float16* A = (const _Float16*)g.A;
#if VAE_HAVE_TDM
        // One TDM descriptor per tile, issued by wave 0 only (EXEC ignored,
        // TENSORcnt tracked per issuing wave). D# per ISA ch.8:
        if (wave == 0) {
            uint32_t lds = (uint32_t)(uintptr_t)&Alds[buf][0];
            uint64_t ga  = (uint64_t)(uintptr_t)(A + (size_t)bM * g.K +
                                                 (size_t)kb * 32);
            u32x4 d0;
            d0[0] = 1u;                                     // count=1, user mode
            d0[1] = lds;                                    // lds_addr
            d0[2] = (uint32_t)ga;                           // global_addr[31:0]
            d0[3] = ((uint32_t)(ga >> 32) & 0x1FFFFFFu)     // global_addr[56:32]
                    | (2u << 30);                           // type=2 (image)
            i32x8 d1;
            d1[0] = (1 << 16)      // data_size = 2 bytes
                  | (1 << 20)      // pad_enable
                  | (3 << 22)      // pad_interval: 16 DWORDs (64B = one row)
                  | (3 << 25);     // pad_amount:   4 DWORDs (16B)
            d1[1] = (int)(((unsigned)g.K & 0xFFFFu) << 16); // tensor_dim0 lo
            d1[2] = (int)(((unsigned)g.K >> 16) & 0xFFFFu)  // tensor_dim0 hi
                  | (128 << 16);                            // tensor_dim1 = 128
            d1[3] = (32 << 16);                             // tile_dim0 = 32
            d1[4] = 128;                                    // tile_dim1 = 128
            d1[5] = g.K;                                    // dim0_stride lo
            d1[6] = 0;                                      // dim0_stride hi
            d1[7] = 0;
            i32x4 dz4 = {0, 0, 0, 0};                       // 2D: groups 2/3
            i32x8 dz8 = {0, 0, 0, 0, 0, 0, 0, 0};
            __builtin_amdgcn_tensor_load_to_lds(d0, d1, dz4, dz4, dz8, 0);
        }
#elif VAE_USE_ASYNC
#pragma unroll
        for (int it = 0; it < 2; ++it) {
            int cid = tid + it * 256;         // 512 chunks of 8 halves
            int row = cid >> 2, cch = cid & 3;
            uint32_t lds = (uint32_t)(uintptr_t)&Alds[buf][row * APITCH + cch * 8];
            uint64_t ga  = (uint64_t)(uintptr_t)(A + (size_t)(bM + row) * g.K +
                                                 (size_t)kb * 32 + cch * 8);
            asm volatile("global_load_async_to_lds_b128 %0, %1, off"
                         :: "v"(lds), "v"(ga) : "memory");
        }
#else
#pragma unroll
        for (int it = 0; it < 2; ++it) {
            int cid = tid + it * 256;
            int row = cid >> 2, cch = cid & 3;
            half8 v = *(const half8*)(A + (size_t)(bM + row) * g.K +
                                      (size_t)kb * 32 + cch * 8);
            *(half8*)&Alds[buf][row * APITCH + cch * 8] = v;
        }
#endif
    };
    auto drainA16 = [&]() {
#if VAE_HAVE_TDM
        if (wave == 0) __builtin_amdgcn_s_wait_tensorcnt(0);
#elif VAE_USE_ASYNC
        asm volatile("s_wait_asynccnt 0x0" ::: "memory");
#endif
    };
    auto loadA_f32 = [&](int kb, f32x4* areg) {
        const float* A = (const float*)g.A;
#pragma unroll
        for (int it = 0; it < 4; ++it) {
            int cid = tid + it * 256;         // 1024 chunks of 4 floats
            int row = cid >> 3, cch = cid & 7;
            areg[it] = *(const f32x4*)(A + (size_t)(bM + row) * g.K +
                                       (size_t)kb * 32 + cch * 4);
        }
    };
    auto commitA_f32 = [&](int buf, const f32x4* areg) {
#pragma unroll
        for (int it = 0; it < 4; ++it) {
            int cid = tid + it * 256;
            int row = cid >> 3, cch = cid & 7;
            half4 h;
#pragma unroll
            for (int e = 0; e < 4; ++e) h[e] = (_Float16)areg[it][e];
            *(half4*)&Alds[buf][row * APITCH + cch * 4] = h;
        }
    };
    // B tile: 32 rows (k) x 128 cols (n), fp32 -> f16, transposed Blds[n][k]
    auto loadB = [&](int kb, f32x4* breg) {
#pragma unroll
        for (int it = 0; it < 4; ++it) {
            int cid = tid + it * 256;         // 1024 chunks of 4 floats
            int k = cid >> 5, nch = cid & 31;
            breg[it] = *(const f32x4*)(g.B + (size_t)(kb * 32 + k) * g.N +
                                       bN + nch * 4);
        }
    };
    auto commitB = [&](int buf, const f32x4* breg) {
#pragma unroll
        for (int it = 0; it < 4; ++it) {
            int cid = tid + it * 256;
            int k = cid >> 5, nch = cid & 31;
#pragma unroll
            for (int e = 0; e < 4; ++e)
                Blds[buf][(nch * 4 + e) * APITCH + k] = (_Float16)breg[it][e];
        }
    };

    // ---- pipelined main loop --------------------------------------------
    const int nk = g.K >> 5;
    f32x4 areg[4], breg[4];

    // prologue: tile 0 -> buffer 0
    if constexpr (AF32) loadA_f32(0, areg); else stageA16(0, 0);
    loadB(0, breg);
    if constexpr (AF32) commitA_f32(0, areg);
    commitB(0, breg);
    if constexpr (!AF32) drainA16();
    __syncthreads();

    for (int kb = 0; kb < nk; ++kb) {
        const int  p    = kb & 1;
        const bool more = (kb + 1 < nk);
        if (more) {
            if constexpr (AF32) loadA_f32(kb + 1, areg); else stageA16(kb + 1, p ^ 1);
            loadB(kb + 1, breg);
        }

        // fragments per ISA 7.12.2 16-bit layouts
        v16h af[2], bf[4];
#pragma unroll
        for (int i = 0; i < 2; ++i) {
            const _Float16* ap = &Alds[p][(wm * 32 + i * 16 + gr) * APITCH];
            half8 lo = *(const half8*)(ap + grp * 8);        // K = g*8 .. g*8+7
            half8 hi = *(const half8*)(ap + 16 + grp * 8);   // K = 16+g*8 ..
            af[i] = pack16(lo, hi);
        }
#pragma unroll
        for (int j = 0; j < 4; ++j) {
            const _Float16* bp = &Blds[p][(wn * 64 + j * 16 + gr) * APITCH + grp * 8];
            half8 lo = *(const half8*)(bp);                  // K = g*8 .. g*8+7
            half8 hi = *(const half8*)(bp + 16);
            bf[j] = pack16(lo, hi);
        }

        if (more) {
            if constexpr (AF32) commitA_f32(p ^ 1, areg);
            commitB(p ^ 1, breg);
        }

#pragma unroll
        for (int i = 0; i < 2; ++i)
#pragma unroll
            for (int j = 0; j < 4; ++j)
                acc[i][j] = __builtin_amdgcn_wmma_f32_16x16x32_f16(
                    false, af[i], false, bf[j], (short)0, acc[i][j], false, false);

        if (more) {
            if constexpr (!AF32) drainA16();
            __syncthreads();
        }
    }

    // ---- epilogue --------------------------------------------------------
    float bsum = 0.0f;
#pragma unroll
    for (int i = 0; i < 2; ++i) {
#pragma unroll
        for (int j = 0; j < 4; ++j) {
            const int n  = bN + wn * 64 + j * 16 + gr;
            const int m0 = bM + wm * 32 + i * 16 + grp * 8;
            const float bv = g.bias[n];
#pragma unroll
            for (int r = 0; r < 8; ++r) {
                const int m = m0 + r;
                float x = acc[i][j][r] + bv;
                if constexpr (EPI == EP_RELU_H16) {
                    x = fmaxf(x, 0.0f);
                    ((_Float16*)g.out)[(size_t)m * g.N + n] = (_Float16)x;
                } else if constexpr (EPI == EP_FEA) {
                    x = fmaxf(x, 0.0f) * g.wcol[m * 2 + g.v];
                    ((float*)g.out)[((size_t)m * g.N + n) * 2 + g.v] = x;
                } else if constexpr (EPI == EP_F32) {
                    ((float*)g.out)[(size_t)m * g.N + n] = x;
                } else { // EP_BCE
                    const float wv = g.wcol[m * 2 + g.v];
                    const float p  = wv / (1.0f + __expf(-x));
                    const float t  = g.target[(size_t)m * g.N + n];
                    const float lp = fmaxf(__logf(p), -100.0f);
                    const float lq = fmaxf(__logf(1.0f - p), -100.0f);
                    bsum -= t * lp + (1.0f - t) * lq;
                }
            }
        }
    }
    if constexpr (EPI == EP_BCE) {
        __shared__ float red[256];
        red[tid] = bsum;
        __syncthreads();
        for (int s = 128; s > 0; s >>= 1) {
            if (tid < s) red[tid] += red[tid + s];
            __syncthreads();
        }
        if (tid == 0) g.partials[blockIdx.y * gridDim.x + blockIdx.x] = red[0];
    }
}

// fusion = (fea0 + fea1) / (w0 + w1), emitted as f16 for the next GEMM
__global__ __launch_bounds__(256) void vae_fuse_kernel(const float* __restrict__ fea,
                                                       const float* __restrict__ w,
                                                       _Float16* __restrict__ fus, int total) {
    int i = blockIdx.x * 256 + threadIdx.x;
    if (i >= total) return;
    int m = i >> 9;  // / H2 (512)
    float f0 = fea[(size_t)i * 2 + 0];
    float f1 = fea[(size_t)i * 2 + 1];
    float ws = w[m * 2 + 0] + w[m * 2 + 1];
    fus[i] = (_Float16)((f0 + f1) / ws);
}

// z = mu + exp(0.5*logvar)*eps ; KL partials per block (deterministic)
__global__ __launch_bounds__(256) void vae_z_kl_kernel(const float* __restrict__ zm,
                                                       const float* __restrict__ zl,
                                                       const float* __restrict__ eps,
                                                       _Float16* __restrict__ z,
                                                       float* __restrict__ partials, int total) {
    const int tid = threadIdx.x;
    int base = (blockIdx.x * 256 + tid) * 8;
    float s = 0.0f;
#pragma unroll
    for (int e = 0; e < 8; ++e) {
        int i = base + e;
        if (i < total) {
            float mu = zm[i], lv = zl[i];
            z[i] = (_Float16)(mu + __expf(0.5f * lv) * eps[i]);
            s += 0.5f * (mu * mu + __expf(lv) - 1.0f - lv);
        }
    }
    __shared__ float red[256];
    red[tid] = s;
    __syncthreads();
    for (int st = 128; st > 0; st >>= 1) {
        if (tid < st) red[tid] += red[tid + st];
        __syncthreads();
    }
    if (tid == 0) partials[blockIdx.x] = red[0];
}

// Fixed-order reduction of partials -> scalar losses (deterministic)
__global__ __launch_bounds__(256) void vae_finalize_kernel(const float* __restrict__ p,
                                                           float* __restrict__ out) {
    __shared__ float red[256];
    const int tid = threadIdx.x;
    float acc[3] = {0.0f, 0.0f, 0.0f};
    for (int a = 0; a < 3; ++a)
        for (int i = tid; i < 1024; i += 256) acc[a] += p[a * 1024 + i];
    float res[3];
    for (int a = 0; a < 3; ++a) {
        red[tid] = acc[a];
        __syncthreads();
        for (int s = 128; s > 0; s >>= 1) {
            if (tid < s) red[tid] += red[tid + s];
            __syncthreads();
        }
        res[a] = red[0];
        __syncthreads();
    }
    if (tid == 0) {
        out[0] = (res[0] + res[1]) * (0.5f / ((float)NROWS * (float)D_IN)); // loss_rec
        out[1] = res[2] / (float)NROWS;                                      // loss_kl
    }
}

// ---- workspace layout (bytes, 256-aligned) ----
#define OFF_H1   ((size_t)0)           // f16 [8192x1024]  16 MiB (reused per view)
#define OFF_FUS  ((size_t)16777216)    // f16 [8192x512]    8 MiB
#define OFF_ZM   ((size_t)25165824)    // f32 [8192x256]    8 MiB
#define OFF_ZL   ((size_t)33554432)    // f32 [8192x256]    8 MiB
#define OFF_Z16  ((size_t)41943040)    // f16 [8192x256]    4 MiB
#define OFF_D1   ((size_t)46137344)    // f16 [8192x512]    8 MiB (reused per view)
#define OFF_D2   ((size_t)54525952)    // f16 [8192x1024]  16 MiB (reused per view)
#define OFF_PART ((size_t)71303168)    // f32 [3*1024] bce0|bce1|kl

extern "C" void kernel_launch(void* const* d_in, const int* in_sizes, int n_in,
                              void* d_out, int out_size, void* d_ws, size_t ws_size,
                              hipStream_t stream) {
    const float* x[2] = {(const float*)d_in[0], (const float*)d_in[1]};
    const float* w    = (const float*)d_in[2];
    const float* eps  = (const float*)d_in[3];
    auto P = [&](int i) { return (const float*)d_in[4 + i]; };
    // per view v (base 10*v): 0 encW0 1 encB0 2 encW1 3 encB1
    //                         4 decW0 5 decB0 6 decW1 7 decB1 8 decW2 9 decB2
    const float* Wm = P(20); const float* bm = P(21);
    const float* Wv = P(22); const float* bv = P(23);

    float* out = (float*)d_out;
    char*  ws  = (char*)d_ws;
    _Float16* h1  = (_Float16*)(ws + OFF_H1);
    _Float16* fus = (_Float16*)(ws + OFF_FUS);
    float*    zm  = (float*)(ws + OFF_ZM);
    float*    zl  = (float*)(ws + OFF_ZL);
    _Float16* z16 = (_Float16*)(ws + OFF_Z16);
    _Float16* d1  = (_Float16*)(ws + OFF_D1);
    _Float16* d2  = (_Float16*)(ws + OFF_D2);
    float*    part = (float*)(ws + OFF_PART); // [bce0 1024][bce1 1024][kl 1024]

    const dim3 blk(256);
    const dim3 gEnc0(H1 / 128, NROWS / 128);   //  8 x 64
    const dim3 gEnc1(H2 / 128, NROWS / 128);   //  4 x 64
    const dim3 gLat(LAT / 128, NROWS / 128);   //  2 x 64
    const dim3 gDec0(H2 / 128, NROWS / 128);   //  4 x 64
    const dim3 gDec1(H1 / 128, NROWS / 128);   //  8 x 64
    const dim3 gDec2(D_IN / 128, NROWS / 128); // 16 x 64 -> 1024 partials

    // ---- encode both views; fea written (scaled by w) straight into d_out ----
    for (int v = 0; v < 2; ++v) {
        GemmArgs a0{};
        a0.A = x[v]; a0.B = P(10 * v + 0); a0.bias = P(10 * v + 1);
        a0.M = NROWS; a0.N = H1; a0.K = D_IN; a0.out = h1;
        vae_gemm_wmma<EP_RELU_H16, true><<<gEnc0, blk, 0, stream>>>(a0);

        GemmArgs a1{};
        a1.A = h1; a1.B = P(10 * v + 2); a1.bias = P(10 * v + 3);
        a1.M = NROWS; a1.N = H2; a1.K = H1; a1.out = out + 2; a1.wcol = w; a1.v = v;
        vae_gemm_wmma<EP_FEA, false><<<gEnc1, blk, 0, stream>>>(a1);
    }

    // ---- fusion ----
    vae_fuse_kernel<<<(NROWS * H2 + 255) / 256, blk, 0, stream>>>(out + 2, w, fus, NROWS * H2);

    // ---- z_mean / z_logvar ----
    GemmArgs am{};
    am.A = fus; am.B = Wm; am.bias = bm; am.M = NROWS; am.N = LAT; am.K = H2; am.out = zm;
    vae_gemm_wmma<EP_F32, false><<<gLat, blk, 0, stream>>>(am);
    GemmArgs av{};
    av.A = fus; av.B = Wv; av.bias = bv; av.M = NROWS; av.N = LAT; av.K = H2; av.out = zl;
    vae_gemm_wmma<EP_F32, false><<<gLat, blk, 0, stream>>>(av);

    // ---- z + KL partials ----
    vae_z_kl_kernel<<<1024, blk, 0, stream>>>(zm, zl, eps, z16, part + 2048, NROWS * LAT);

    // ---- decode both views, fused sigmoid+BCE on the last GEMM ----
    for (int v = 0; v < 2; ++v) {
        GemmArgs b0{};
        b0.A = z16; b0.B = P(10 * v + 4); b0.bias = P(10 * v + 5);
        b0.M = NROWS; b0.N = H2; b0.K = LAT; b0.out = d1;
        vae_gemm_wmma<EP_RELU_H16, false><<<gDec0, blk, 0, stream>>>(b0);

        GemmArgs b1{};
        b1.A = d1; b1.B = P(10 * v + 6); b1.bias = P(10 * v + 7);
        b1.M = NROWS; b1.N = H1; b1.K = H2; b1.out = d2;
        vae_gemm_wmma<EP_RELU_H16, false><<<gDec1, blk, 0, stream>>>(b1);

        GemmArgs b2{};
        b2.A = d2; b2.B = P(10 * v + 8); b2.bias = P(10 * v + 9);
        b2.M = NROWS; b2.N = D_IN; b2.K = H1;
        b2.wcol = w; b2.v = v; b2.target = x[v]; b2.partials = part + v * 1024;
        vae_gemm_wmma<EP_BCE, false><<<gDec2, blk, 0, stream>>>(b2);
    }

    // ---- scalars ----
    vae_finalize_kernel<<<1, blk, 0, stream>>>(part, out);
}
